// SpectralMultiHeadAttention_69363721830452
// MI455X (gfx1250) — compile-verified, matrix operand order checked by reading, not run on hardware
//
#include <hip/hip_runtime.h>
#include <hip/hip_bf16.h>

// SpectralMultiHeadAttention for MI455X (gfx1250, wave32, WMMA + TDM + async).
//
// Algebra: attn contracts over hw (16384), so
//   G_b   = xs_bᵀ xs_b                     (256x256 Gram per batch)
//   attn  = softmax( norm(Wq_hᵀ G Wk_h) )  (tiny, 8 x 32x32)
//   M_b   = Wv · blockdiag(attn_hᵀ) · Wout (256x256)
//   out_b = xs_b · M_b
// Two streaming GEMM passes over x in bf16 WMMA (fp32 accumulate).
// Operand tiles staged to LDS with the Tensor Data Mover (TENSOR_LOAD_TO_LDS,
// TENSORcnt) and GLOBAL_LOAD_ASYNC_TO_LDS_B128 (ASYNCcnt); split-K Gram
// accumulates with f32 global atomics.

typedef __attribute__((ext_vector_type(16))) __bf16    v16bf;
typedef __attribute__((ext_vector_type(8)))  float     v8f;
typedef __attribute__((ext_vector_type(4)))  unsigned  u32x4;
typedef __attribute__((ext_vector_type(8)))  unsigned  u32x8;

constexpr int CC     = 256;     // channels
constexpr int HW     = 16384;   // 128*128 spatial positions
constexpr int NB     = 4;       // batch
constexpr int HEADS  = 8;
constexpr int KCHUNK = 2048;    // split-K chunk for the Gram pass (8 chunks)

// ---------------------------------------------------------------------------
// CDNA5 async global->LDS copy (16 bytes per op, per lane), ASYNCcnt.
// ---------------------------------------------------------------------------
__device__ __forceinline__ void async_copy16(unsigned lds_byte_addr, const void* gptr) {
  const unsigned long long ga = (unsigned long long)gptr;
  asm volatile("global_load_async_to_lds_b128 %0, %1, off"
               :: "v"(lds_byte_addr), "v"(ga)
               : "memory");
}

__device__ __forceinline__ void wait_async0() {
#if __has_builtin(__builtin_amdgcn_s_wait_asynccnt)
  __builtin_amdgcn_s_wait_asynccnt(0);
#else
  asm volatile("s_wait_asynccnt 0x0" ::: "memory");
#endif
}

__device__ __forceinline__ void wait_tensor0() {
#if __has_builtin(__builtin_amdgcn_s_wait_tensorcnt)
  __builtin_amdgcn_s_wait_tensorcnt(0);
#else
  asm volatile("s_wait_tensorcnt 0x0" ::: "memory");
#endif
}

// ---------------------------------------------------------------------------
// Tensor Data Mover: load a 2D tile (tile_w x tile_h elems, 4B elems) of a
// row-major tensor (row stride = stride_elems) into LDS.  D# built per ISA
// CDNA5 ISA; groups 2/3 omitted (<=2D tensor).
// All inputs must be wave-uniform (descriptor lives in SGPRs).
// ---------------------------------------------------------------------------
__device__ __forceinline__ void tdm_load_2d_f32(unsigned lds_byte, const void* gaddr,
                                                unsigned tile_w, unsigned tile_h,
                                                unsigned tensor_w, unsigned tensor_h,
                                                unsigned stride_elems) {
  const unsigned long long ga = (unsigned long long)gaddr;
  u32x4 g0;
  g0[0] = 1u;                                            // count=1, user mode
  g0[1] = lds_byte;                                      // lds_addr
  g0[2] = (unsigned)ga;                                  // global_addr[31:0]
  g0[3] = (unsigned)((ga >> 32) & 0x01FFFFFFu)           // global_addr[56:32]
          | 0x80000000u;                                 // type=2 ("image")
  u32x8 g1;
  g1[0] = 2u << 16;                                      // data_size=2 (4 bytes)
  g1[1] = (tensor_w & 0xFFFFu) << 16;                    // tensor_dim0[15:0]
  g1[2] = ((tensor_w >> 16) & 0xFFFFu)                   // tensor_dim0[31:16]
          | ((tensor_h & 0xFFFFu) << 16);                // tensor_dim1[15:0]
  g1[3] = ((tensor_h >> 16) & 0xFFFFu)                   // tensor_dim1[31:16]
          | ((tile_w & 0xFFFFu) << 16);                  // tile_dim0
  g1[4] = tile_h & 0xFFFFu;                              // tile_dim1 (tile_dim2=0)
  g1[5] = stride_elems;                                  // tensor_dim0_stride[31:0]
  g1[6] = 0u;
  g1[7] = 0u;
  asm volatile("tensor_load_to_lds %0, %1" :: "s"(g0), "s"(g1) : "memory");
}

// LDS byte offset of a __shared__ object (LDS aperture: low 32 bits = offset).
__device__ __forceinline__ unsigned lds_off(const void* p) {
  return (unsigned)(size_t)p;
}

// ---------------------------------------------------------------------------
// WMMA fragment builders (16x16x32 bf16). Per the CDNA5 ISA layout tables:
//  A (16x32, MxK):  lane L -> m = L&15, kbase = (L>>4)*8,
//                   element e -> K = kbase + (e&7) + 16*(e>>3)
//  B (32x16, KxN):  mirrored (n = L&15, same K pattern)
//  C/D (16x16 f32): element v -> m = (L>>4)*8 + v, n = L&15
// ---------------------------------------------------------------------------

// K-major fragment gathered from an LDS tile (row stride `ld` floats).
__device__ __forceinline__ v16bf frag_lds_kmajor(const float* tile, int ld,
                                                 int colbase, int lane) {
  const int mn = lane & 15;
  const int kb = (lane >> 4) << 3;
  const float* p = tile + colbase + mn;
  v16bf f;
#pragma unroll
  for (int e = 0; e < 16; ++e) {
    const int k = kb + (e & 7) + ((e >> 3) << 4);
    f[e] = (__bf16)p[k * ld];
  }
  return f;
}

// M-major fragment from global memory, vectorized: each lane reads two
// contiguous 8-float runs (4x float4) from its row.
__device__ __forceinline__ v16bf frag_mmajor_vec(const float* __restrict__ src,
                                                 int ld, int row0, int k0, int lane) {
  const int m  = lane & 15;
  const int kb = (lane >> 4) << 3;
  const float* r = src + (size_t)(row0 + m) * ld + k0 + kb;
  const float4 u0 = ((const float4*)r)[0];
  const float4 u1 = ((const float4*)r)[1];
  const float4 u2 = ((const float4*)(r + 16))[0];
  const float4 u3 = ((const float4*)(r + 16))[1];
  v16bf f;
  f[0]  = (__bf16)u0.x; f[1]  = (__bf16)u0.y; f[2]  = (__bf16)u0.z; f[3]  = (__bf16)u0.w;
  f[4]  = (__bf16)u1.x; f[5]  = (__bf16)u1.y; f[6]  = (__bf16)u1.z; f[7]  = (__bf16)u1.w;
  f[8]  = (__bf16)u2.x; f[9]  = (__bf16)u2.y; f[10] = (__bf16)u2.z; f[11] = (__bf16)u2.w;
  f[12] = (__bf16)u3.x; f[13] = (__bf16)u3.y; f[14] = (__bf16)u3.z; f[15] = (__bf16)u3.w;
  return f;
}

__device__ __forceinline__ v8f wmma_bf16(v16bf a, v16bf b, v8f c) {
  // (neg_a, A, neg_b, B, c_mod, C, reuse_a, reuse_b)
  return __builtin_amdgcn_wmma_f32_16x16x32_bf16(false, a, false, b, (short)0, c,
                                                 false, false);
}

__device__ __forceinline__ v8f v8f_zero() {
  v8f z = {0.f, 0.f, 0.f, 0.f, 0.f, 0.f, 0.f, 0.f};
  return z;
}

// ---------------------------------------------------------------------------
// Kernel 0: zero the Gram accumulator region (re-zeroed every call: the split-K
// pass accumulates with atomics, and the harness replays the graph).
// ---------------------------------------------------------------------------
__global__ void zero_kernel(float* __restrict__ p, int n) {
  const int i = blockIdx.x * blockDim.x + threadIdx.x;
  if (i < n) p[i] = 0.f;
}

// ---------------------------------------------------------------------------
// Kernel 1: Gram matrices  G_b = xs_bᵀ xs_b  (256x256 f32, bf16 WMMA inputs).
// Grid: x = 8 output tiles (4 row-blocks of 64 x 2 col-blocks of 128),
//       y = 8 K-chunks of 2048, z = batch.  Block: 128 threads = 4 waves.
// Per 32-row K-step: A tile staged with async b128 copies (ASYNCcnt),
// B tile staged with one TDM descriptor per wave (TENSORcnt); bf16 fragments
// gathered from LDS.  Wave tile 32x64 -> 2x4 accumulators; split-K partials
// atomicAdd'ed into G.
// ---------------------------------------------------------------------------
__global__ __launch_bounds__(128) void gram_kernel(const float* __restrict__ x,
                                                   float* __restrict__ G) {
  const int b     = blockIdx.z;
  const int kbase = blockIdx.y * KCHUNK;
  const int it    = blockIdx.x >> 1;   // 0..3 row block (64 wide in c)
  const int jt    = blockIdx.x & 1;    // 0..1 col block (128 wide in c)
  const float* xs = x + (size_t)b * HW * CC;
  float* Gb       = G + (size_t)b * CC * CC;

  const int tid  = threadIdx.x;
  const int wave = tid >> 5;
  const int lane = tid & 31;
  const int i0   = it * 64;                 // global col base of A tile
  const int j0   = jt * 128;                // global col base of B tile
  const int mw   = (wave >> 1) * 32;        // wave row base within A tile
  const int nw   = (wave & 1) * 64;         // wave col base within B tile
  const int wv   = __builtin_amdgcn_readfirstlane(wave);  // SGPR wave id for TDM

  __shared__ __align__(16) float tA[32 * 64];    //  8 KB: rows k, cols i0..i0+63
  __shared__ __align__(16) float tB[32 * 128];   // 16 KB: rows k, cols j0..j0+127
  const unsigned ldsA = lds_off(tA);
  const unsigned ldsB = lds_off(tB);

  v8f acc[2][4];
#pragma unroll
  for (int i = 0; i < 2; ++i)
#pragma unroll
    for (int j = 0; j < 4; ++j) acc[i][j] = v8f_zero();

  for (int kk = 0; kk < KCHUNK; kk += 32) {
    const int k0 = kbase + kk;

    // ---- A tile: 32x64 fp32 = 512 float4, 4 async copies per thread
    //      (fixed trip count: no per-copy EXEC guards)
#pragma unroll
    for (int i = 0; i < 4; ++i) {
      const int t   = tid + i * 128;
      const int row = t >> 4, c4 = t & 15;
      async_copy16(ldsA + (unsigned)t * 16,
                   xs + (size_t)(k0 + row) * CC + i0 + (c4 << 2));
    }
    // ---- B tile: 32x128 fp32 via TDM; each wave moves an 8x128 slice
    {
      const int wrow = wv * 8;
      tdm_load_2d_f32(ldsB + (unsigned)(wrow * 128 * 4),
                      xs + (size_t)(k0 + wrow) * CC + j0,
                      /*tile_w=*/128, /*tile_h=*/8,
                      /*tensor_w=*/CC, /*tensor_h=*/HW, /*stride=*/CC);
    }
    wait_async0();
    wait_tensor0();
    __syncthreads();

    // ---- fragments from LDS, 8 WMMAs
    const v16bf a0 = frag_lds_kmajor(tA, 64,  mw,      lane);
    const v16bf a1 = frag_lds_kmajor(tA, 64,  mw + 16, lane);
    const v16bf b0 = frag_lds_kmajor(tB, 128, nw,      lane);
    const v16bf b1 = frag_lds_kmajor(tB, 128, nw + 16, lane);
    const v16bf b2 = frag_lds_kmajor(tB, 128, nw + 32, lane);
    const v16bf b3 = frag_lds_kmajor(tB, 128, nw + 48, lane);
    acc[0][0] = wmma_bf16(a0, b0, acc[0][0]);
    acc[0][1] = wmma_bf16(a0, b1, acc[0][1]);
    acc[0][2] = wmma_bf16(a0, b2, acc[0][2]);
    acc[0][3] = wmma_bf16(a0, b3, acc[0][3]);
    acc[1][0] = wmma_bf16(a1, b0, acc[1][0]);
    acc[1][1] = wmma_bf16(a1, b1, acc[1][1]);
    acc[1][2] = wmma_bf16(a1, b2, acc[1][2]);
    acc[1][3] = wmma_bf16(a1, b3, acc[1][3]);
    __syncthreads();   // protect LDS tiles before next k-step overwrites
  }

  const int nl = lane & 15;
  const int mb = (lane >> 4) << 3;
#pragma unroll
  for (int mi = 0; mi < 2; ++mi)
#pragma unroll
    for (int ni = 0; ni < 4; ++ni) {
      const int m0 = i0 + mw + mi * 16 + mb;
      const int n  = j0 + nw + ni * 16 + nl;
#pragma unroll
      for (int v = 0; v < 8; ++v)
        atomicAdd(&Gb[(size_t)(m0 + v) * CC + n], acc[mi][ni][v]);
    }
}

// ---------------------------------------------------------------------------
// Kernel 2: per-batch combine (tiny FLOPs, ~0.1 GFLOP/batch, fp32 VALU).
//   Pq = G@Wq, Pk = G@Wk
//   qn[hc] = 1/sqrt(Σ_u Wq[u,hc] Pq[u,hc]);  kn likewise from Pk
//   attn[h,c,e] = softmax_e( Σ_u Wq[u,hc] Pk[u,he] * qn*kn / sqrt(hw) )
//   T[he,o] = Σ_c attn[h,c,e] Wout[hc,o];  M[u,o] = Σ_he Wv[u,he] T[he,o]
// One workgroup (256 threads) per batch; attn + norms live in LDS.
// ---------------------------------------------------------------------------
__global__ __launch_bounds__(256) void attn_combine_kernel(
    const float* __restrict__ G,  const float* __restrict__ Wq,
    const float* __restrict__ Wk, const float* __restrict__ Wv,
    const float* __restrict__ Wout,
    float* __restrict__ Pq, float* __restrict__ Pk,
    float* __restrict__ T,  float* __restrict__ M) {
  const int b   = blockIdx.x;
  const int tid = threadIdx.x;
  const float* Gb = G  + (size_t)b * CC * CC;
  float* Pqb      = Pq + (size_t)b * CC * CC;
  float* Pkb      = Pk + (size_t)b * CC * CC;
  float* Tb       = T  + (size_t)b * CC * CC;
  float* Mb       = M  + (size_t)b * CC * CC;

  __shared__ float attn[HEADS][32][32];
  __shared__ float qn[CC];
  __shared__ float kn[CC];

  // Stage A: Pq = G@Wq, Pk = G@Wk
  for (int idx = tid; idx < CC * CC; idx += 256) {
    const int u = idx >> 8, col = idx & 255;
    const float* gr = Gb + (size_t)u * CC;
    float sq = 0.f, sk = 0.f;
    for (int v = 0; v < CC; ++v) {
      const float g = gr[v];
      sq = fmaf(g, Wq[v * CC + col], sq);
      sk = fmaf(g, Wk[v * CC + col], sk);
    }
    Pqb[idx] = sq;
    Pkb[idx] = sk;
  }
  __threadfence();
  __syncthreads();

  // Stage B: inverse L2 norms of q/k rows
  {
    const int col = tid;  // 256 threads == 256 columns
    float sq = 0.f, sk = 0.f;
    for (int u = 0; u < CC; ++u) {
      sq = fmaf(Wq[u * CC + col], Pqb[u * CC + col], sq);
      sk = fmaf(Wk[u * CC + col], Pkb[u * CC + col], sk);
    }
    qn[col] = rsqrtf(sq);
    kn[col] = rsqrtf(sk);
  }
  __syncthreads();

  // Stage C: attn logits (scaled by 1/sqrt(d_k)=1/128 since d_k = hw = 16384)
  for (int idx = tid; idx < HEADS * 32 * 32; idx += 256) {
    const int h = idx >> 10, ce = idx & 1023, c = ce >> 5, e = ce & 31;
    const int hc = h * 32 + c, he = h * 32 + e;
    float s = 0.f;
    for (int u = 0; u < CC; ++u) s = fmaf(Wq[u * CC + hc], Pkb[u * CC + he], s);
    attn[h][c][e] = s * qn[hc] * kn[he] * 0.0078125f;
  }
  __syncthreads();

  // Stage D: row softmax (one 32-wide row per thread)
  {
    const int h = tid >> 5, c = tid & 31;
    float mx = -3.402823466e38f;
    for (int e = 0; e < 32; ++e) mx = fmaxf(mx, attn[h][c][e]);
    float s = 0.f;
    for (int e = 0; e < 32; ++e) {
      const float t = __expf(attn[h][c][e] - mx);
      attn[h][c][e] = t;
      s += t;
    }
    const float inv = 1.0f / s;
    for (int e = 0; e < 32; ++e) attn[h][c][e] *= inv;
  }
  __syncthreads();

  // Stage E: T[h*32+e, o] = Σ_c attn[h,c,e] * Wout[h*32+c, o]
  for (int idx = tid; idx < CC * CC; idx += 256) {
    const int he = idx >> 8, o = idx & 255, h = he >> 5, e = he & 31;
    float s = 0.f;
    for (int c = 0; c < 32; ++c)
      s = fmaf(attn[h][c][e], Wout[(h * 32 + c) * CC + o], s);
    Tb[idx] = s;
  }
  __threadfence();
  __syncthreads();

  // Stage F: M[u,o] = Σ_he Wv[u,he] * T[he,o]
  for (int idx = tid; idx < CC * CC; idx += 256) {
    const int u = idx >> 8, o = idx & 255;
    const float* wr = Wv + (size_t)u * CC;
    float s = 0.f;
    for (int he = 0; he < CC; ++he) s = fmaf(wr[he], Tb[he * CC + o], s);
    Mb[idx] = s;
  }
}

// ---------------------------------------------------------------------------
// Kernel 3: out_b = xs_b @ M_b  (16384x256 @ 256x256 per batch, bf16 WMMA).
// Grid: x = 128 row tiles (128 hw rows), y = 2 col tiles (128 c cols),
//       z = batch.  Block: 256 threads = 8 waves; wave tile 32x64 (2x4 accs).
// B tile (M matrix, shared by all 8 waves) staged to LDS via TDM (each wave
// moves a 4x128 slice); A rows streamed from global with float4 loads +
// prefetch.
// ---------------------------------------------------------------------------
__global__ __launch_bounds__(256) void out_gemm_kernel(const float* __restrict__ x,
                                                       const float* __restrict__ M,
                                                       float* __restrict__ out) {
  const int b  = blockIdx.z;
  const int rt = blockIdx.x;  // 128 row tiles
  const int ct = blockIdx.y;  // 2 col tiles
  const float* xs = x   + (size_t)b * HW * CC;
  const float* Mb = M   + (size_t)b * CC * CC;
  float* ob       = out + (size_t)b * HW * CC;

  const int tid  = threadIdx.x;
  const int wave = tid >> 5;
  const int lane = tid & 31;
  const int mw   = rt * 128 + (wave >> 1) * 32;  // hw row base
  const int n0   = ct * 128;                     // c col base of the B tile
  const int nw   = (wave & 1) * 64;              // wave col base within B tile
  const int wv   = __builtin_amdgcn_readfirstlane(wave);  // SGPR wave id, 0..7

  __shared__ __align__(16) float tB[32 * 128];   // 16 KB: rows k, cols n0..n0+127
  const unsigned ldsB = lds_off(tB);

  v8f acc[2][4];
#pragma unroll
  for (int i = 0; i < 2; ++i)
#pragma unroll
    for (int j = 0; j < 4; ++j) acc[i][j] = v8f_zero();

#pragma unroll 1
  for (int k0 = 0; k0 < CC; k0 += 32) {
    // ---- B tile: 32x128 fp32 via TDM; each of 8 waves moves a 4x128 slice
    {
      const int wrow = wv * 4;
      tdm_load_2d_f32(ldsB + (unsigned)(wrow * 128 * 4),
                      Mb + (size_t)(k0 + wrow) * CC + n0,
                      /*tile_w=*/128, /*tile_h=*/4,
                      /*tensor_w=*/CC, /*tensor_h=*/CC, /*stride=*/CC);
    }
    wait_tensor0();
    __syncthreads();

    __builtin_prefetch(xs + (size_t)(mw + lane) * CC + k0 + 32, 0, 1);
    const v16bf a0 = frag_mmajor_vec(xs, CC, mw,      k0, lane);
    const v16bf a1 = frag_mmajor_vec(xs, CC, mw + 16, k0, lane);
    const v16bf b0 = frag_lds_kmajor(tB, 128, nw,      lane);
    const v16bf b1 = frag_lds_kmajor(tB, 128, nw + 16, lane);
    const v16bf b2 = frag_lds_kmajor(tB, 128, nw + 32, lane);
    const v16bf b3 = frag_lds_kmajor(tB, 128, nw + 48, lane);
    acc[0][0] = wmma_bf16(a0, b0, acc[0][0]);
    acc[0][1] = wmma_bf16(a0, b1, acc[0][1]);
    acc[0][2] = wmma_bf16(a0, b2, acc[0][2]);
    acc[0][3] = wmma_bf16(a0, b3, acc[0][3]);
    acc[1][0] = wmma_bf16(a1, b0, acc[1][0]);
    acc[1][1] = wmma_bf16(a1, b1, acc[1][1]);
    acc[1][2] = wmma_bf16(a1, b2, acc[1][2]);
    acc[1][3] = wmma_bf16(a1, b3, acc[1][3]);
    __syncthreads();   // protect LDS tile before next k-step overwrites
  }

  const int nl = lane & 15;
  const int mb = (lane >> 4) << 3;
#pragma unroll
  for (int mi = 0; mi < 2; ++mi)
#pragma unroll
    for (int ni = 0; ni < 4; ++ni) {
      const int m0 = mw + mi * 16 + mb;
      const int n  = n0 + nw + ni * 16 + nl;
#pragma unroll
      for (int v = 0; v < 8; ++v)
        ob[(size_t)(m0 + v) * CC + n] = acc[mi][ni][v];
    }
}

// ---------------------------------------------------------------------------
// Launch
// ---------------------------------------------------------------------------
extern "C" void kernel_launch(void* const* d_in, const int* in_sizes, int n_in,
                              void* d_out, int out_size, void* d_ws, size_t ws_size,
                              hipStream_t stream) {
  const float* x    = (const float*)d_in[0];
  const float* Wq   = (const float*)d_in[1];
  const float* Wk   = (const float*)d_in[2];
  const float* Wv   = (const float*)d_in[3];
  const float* Wout = (const float*)d_in[4];
  float* out = (float*)d_out;

  // Workspace layout (floats): G | Pq | Pk | T | M  (each 4 * 256*256)
  float* ws = (float*)d_ws;
  const size_t seg = (size_t)NB * CC * CC;  // 262144 floats
  float* G  = ws;
  float* Pq = ws + seg;
  float* Pk = ws + 2 * seg;
  float* T  = ws + 3 * seg;
  float* M  = ws + 4 * seg;

  // 0) zero split-K accumulator
  zero_kernel<<<dim3((unsigned)((seg + 255) / 256)), dim3(256), 0, stream>>>(G, (int)seg);

  // 1) Gram: G_b = xs_bᵀ xs_b   (bf16 WMMA, TDM + async staging, split-K atomics)
  gram_kernel<<<dim3(8, HW / KCHUNK, NB), dim3(128), 0, stream>>>(x, G);

  // 2) attention combine -> M_b
  attn_combine_kernel<<<dim3(NB), dim3(256), 0, stream>>>(G, Wq, Wk, Wv, Wout,
                                                          Pq, Pk, T, M);

  // 3) out_b = xs_b @ M_b  (bf16 WMMA, TDM-staged B tile)
  out_gemm_kernel<<<dim3(HW / 128, CC / 128, NB), dim3(256), 0, stream>>>(x, M, out);
}
